// Correlation_8435315769394
// MI455X (gfx1250) — matrix operand order repaired and checked by reading, not compiled, not run on hardware
//
#include <hip/hip_runtime.h>

// CDNA5 (gfx1250) 3D correlation / cost-volume.
// out[(di*9+dj)*9+dk, h, w, d] = (1/32) * sum_c x1[c,h,w,d] * x2[c,h+di-4,w+dj-4,d+dk-4]
//
// Fast path: split fp32 into hi/lo f16 (prepass into d_ws, position-major layout),
// compute banded Gram matrices G[m,n] = sum_c x2row[c,m]*x1row[c,n] with
// V_WMMA_F32_16X16X32_F16 (3 WMMAs per 16x16 tile: hh + hl + lh, ~2^-22 rel. error),
// then extract the 9-wide band. Fallback: exact V_WMMA_F32_16X16X4_F32 path.

typedef __attribute__((ext_vector_type(2)))  float    v2f;
typedef __attribute__((ext_vector_type(8)))  float    v8f;
typedef __attribute__((ext_vector_type(8)))  _Float16 v8h;
typedef __attribute__((ext_vector_type(16))) _Float16 v16h;

#define CSTRIDE 110592  /* 48*48*48 : channel stride (x1/x2) and offset stride (out) */
#define HSTRIDE 2304    /* 48*48 */
#define WSTRIDE 48
#define NPOS    110592  /* flat spatial positions */
#define TPLANE  3538944 /* NPOS*32 : one split plane in f16 elements */

// ---------------------------------------------------------------------------
// Prepass: split fp32 -> (hi,lo) f16 and transpose [c][p] -> [p][c].
// grid = (432, 2): blockIdx.y selects x1 / x2. 256 threads handle 256 positions.
// ws layout (f16): [0]=x1h, [1]=x1l, [2]=x2h, [3]=x2l, each TPLANE elements.
// ---------------------------------------------------------------------------
__global__ __launch_bounds__(256) void corr_split_kernel(
    const float* __restrict__ x1, const float* __restrict__ x2,
    _Float16* __restrict__ ws)
{
    const float* src = blockIdx.y ? x2 : x1;
    _Float16* dh = ws + (size_t)blockIdx.y * 2 * TPLANE;
    _Float16* dl = dh + TPLANE;

    __shared__ alignas(16) _Float16 Lh[256][40];  // pad 32->40 (bank spread, 16B rows)
    __shared__ alignas(16) _Float16 Ll[256][40];

    const int tid   = threadIdx.x;
    const int pbase = blockIdx.x * 256;

#pragma unroll 4
    for (int c = 0; c < 32; ++c) {
        float v = src[c * CSTRIDE + pbase + tid];       // coalesced b32
        _Float16 h = (_Float16)v;
        _Float16 l = (_Float16)(v - (float)h);
        Lh[tid][c] = h;
        Ll[tid][c] = l;
    }
    __syncthreads();

    // Write out 256 pos * 32 ch = 8192 f16 per array as coalesced b128 stores.
#pragma unroll
    for (int r = 0; r < 4; ++r) {
        const int q    = r * 256 + tid;     // 16-byte chunk index
        const int pos  = q >> 2;
        const int part = q & 3;
        v8h hh = *(const v8h*)&Lh[pos][part * 8];
        v8h ll = *(const v8h*)&Ll[pos][part * 8];
        *(v8h*)(dh + (size_t)pbase * 32 + (size_t)q * 8) = hh;
        *(v8h*)(dl + (size_t)pbase * 32 + (size_t)q * 8) = ll;
    }
}

// Build a v16h A-operand fragment (16x32 f16 A layout):
// lanes<16 need channels {half*8..+7} and {16+half*8..+7} at one position.
__device__ __forceinline__ v16h load_fragA(const _Float16* posbase, int half) {
    v8h a = *(const v8h*)(posbase + half * 8);
    v8h b = *(const v8h*)(posbase + 16 + half * 8);
    return __builtin_shufflevector(a, b, 0, 1, 2, 3, 4, 5, 6, 7,
                                         8, 9, 10, 11, 12, 13, 14, 15);
}

// Build a v16h B-operand fragment (32x16 f16 B layout):
// lanes<16 hold K(=channel) {half*8..+7} and {16+half*8..+7}, column n = lane%16.
__device__ __forceinline__ v16h load_fragB(const _Float16* posbase, int half) {
    v8h a = *(const v8h*)(posbase + half * 8);
    v8h b = *(const v8h*)(posbase + 16 + half * 8);
    return __builtin_shufflevector(a, b, 0, 1, 2, 3, 4, 5, 6, 7,
                                         8, 9, 10, 11, 12, 13, 14, 15);
}

// ---------------------------------------------------------------------------
// Main f16-split kernel: one block per (h,w); 8 wave32s split the 81 (di,dj).
// ---------------------------------------------------------------------------
__global__ __launch_bounds__(256) void corr3d_wmma_f16_kernel(
    const _Float16* __restrict__ ws, float* __restrict__ out)
{
    const _Float16* x1h = ws;
    const _Float16* x1l = ws + TPLANE;
    const _Float16* x2h = ws + 2 * TPLANE;
    const _Float16* x2l = ws + 3 * TPLANE;

    const int hw   = blockIdx.x;
    const int h    = hw / 48;
    const int w    = hw - h * 48;
    const int tid  = threadIdx.x;
    const int wave = tid >> 5;
    const int lane = tid & 31;
    const int half = lane >> 4;
    const int lid  = lane & 15;

    __shared__ float G[8][2][16][16];

    const int hwpos = (h * 48 + w) * 48;   // flat position of (h,w,0)
    const int hwoff = h * HSTRIDE + w * WSTRIDE;

    // Hoisted B operands (x1 row), reused across all 81 pairs.
    v16h Bh[3], Bl[3];
#pragma unroll
    for (int t = 0; t < 3; ++t) {
        const _Float16* bp_h = x1h + (size_t)(hwpos + t * 16 + lid) * 32;
        const _Float16* bp_l = x1l + (size_t)(hwpos + t * 16 + lid) * 32;
        Bh[t] = load_fragB(bp_h, half);
        Bl[t] = load_fragB(bp_l, half);
    }

    for (int p = wave; p < 81; p += 8) {
        const int di = p / 9;
        const int dj = p - di * 9;
        const int h2 = h + di - 4;
        const int w2 = w + dj - 4;
        const bool valid = ((unsigned)h2 < 48u) && ((unsigned)w2 < 48u);

        v8f acc[6];
#pragma unroll
        for (int q = 0; q < 6; ++q) { v8f z = {}; acc[q] = z; }

        if (valid) {
            const int p2 = (h2 * 48 + w2) * 48;
            // 4 overlapping 16-wide m-windows along D (zero-masked padding).
            v16h Ah[4], Al[4];
#pragma unroll
            for (int j = 0; j < 4; ++j) {
                const int m = j * 16 - 4 + lid;
                v16h zh = {}; v16h zl = {};
                if ((unsigned)m < 48u) {
                    const _Float16* ab = x2h + (size_t)(p2 + m) * 32;
                    const _Float16* lb = x2l + (size_t)(p2 + m) * 32;
                    zh = load_fragA(ab, half);
                    zl = load_fragA(lb, half);
                }
                Ah[j] = zh;
                Al[j] = zl;
            }
            // n-tile t uses m-windows t and t+1; 3 WMMAs per Gram tile (hh,hl,lh).
#pragma unroll
            for (int t = 0; t < 3; ++t) {
                acc[2 * t] = __builtin_amdgcn_wmma_f32_16x16x32_f16(
                    false, Ah[t], false, Bh[t], (short)0, acc[2 * t], false, false);
                acc[2 * t + 1] = __builtin_amdgcn_wmma_f32_16x16x32_f16(
                    false, Ah[t + 1], false, Bh[t], (short)0, acc[2 * t + 1], false, false);
                acc[2 * t] = __builtin_amdgcn_wmma_f32_16x16x32_f16(
                    false, Ah[t], false, Bl[t], (short)0, acc[2 * t], false, false);
                acc[2 * t + 1] = __builtin_amdgcn_wmma_f32_16x16x32_f16(
                    false, Ah[t + 1], false, Bl[t], (short)0, acc[2 * t + 1], false, false);
                acc[2 * t] = __builtin_amdgcn_wmma_f32_16x16x32_f16(
                    false, Al[t], false, Bh[t], (short)0, acc[2 * t], false, false);
                acc[2 * t + 1] = __builtin_amdgcn_wmma_f32_16x16x32_f16(
                    false, Al[t + 1], false, Bh[t], (short)0, acc[2 * t + 1], false, false);
            }
        }

        // Band extraction: out[p*9+dk][h][w][t*16+n] = G[n+dk, n] / 32
#pragma unroll
        for (int t = 0; t < 3; ++t) {
#pragma unroll
            for (int r = 0; r < 8; ++r) {
                G[wave][0][r + half * 8][lid] = acc[2 * t][r];
                G[wave][1][r + half * 8][lid] = acc[2 * t + 1][r];
            }
            asm volatile("s_wait_dscnt 0x0" ::: "memory");

            const int obase = (p * 9) * CSTRIDE + hwoff + t * 16 + lid;
#pragma unroll
            for (int dkp = 0; dkp < 5; ++dkp) {
                const int dk = dkp * 2 + half;
                if (dk < 9) {
                    const int idx = lid + dk;
                    float v = (idx < 16) ? G[wave][0][idx][lid]
                                         : G[wave][1][idx - 16][lid];
                    out[obase + dk * CSTRIDE] = v * (1.0f / 32.0f);
                }
            }
            asm volatile("s_wait_dscnt 0x0" ::: "memory");
        }
    }
}

// ---------------------------------------------------------------------------
// Fallback: exact fp32 WMMA path (V_WMMA_F32_16X16X4_F32), no workspace needed.
// ---------------------------------------------------------------------------
__global__ __launch_bounds__(256) void corr3d_wmma_f32_kernel(
    const float* __restrict__ x1, const float* __restrict__ x2,
    float* __restrict__ out)
{
    const int hw   = blockIdx.x;
    const int h    = hw / 48;
    const int w    = hw - h * 48;
    const int tid  = threadIdx.x;
    const int wave = tid >> 5;
    const int lane = tid & 31;
    const int half = lane >> 4;
    const int lid  = lane & 15;

    __shared__ float G[8][2][16][16];

    const int hwoff = h * HSTRIDE + w * WSTRIDE;

    v2f Breg[3][8];
#pragma unroll
    for (int t = 0; t < 3; ++t) {
#pragma unroll
        for (int i = 0; i < 8; ++i) {
            const float* bp = x1 + (i * 4 + half * 2) * CSTRIDE + hwoff + t * 16 + lid;
            v2f b; b.x = bp[0]; b.y = bp[CSTRIDE];
            Breg[t][i] = b;
        }
    }

    for (int p = wave; p < 81; p += 8) {
        const int di = p / 9;
        const int dj = p - di * 9;
        const int h2 = h + di - 4;
        const int w2 = w + dj - 4;
        const bool valid = ((unsigned)h2 < 48u) && ((unsigned)w2 < 48u);

        v8f acc[6];
#pragma unroll
        for (int q = 0; q < 6; ++q) { v8f z = {}; acc[q] = z; }

        if (valid) {
            const float* x2base = x2 + h2 * HSTRIDE + w2 * WSTRIDE;
#pragma unroll
            for (int i = 0; i < 8; ++i) {
                const int ca = i * 4 + half * 2;
                v2f A[4];
#pragma unroll
                for (int j = 0; j < 4; ++j) {
                    const int m = j * 16 - 4 + lid;
                    v2f a; a.x = 0.0f; a.y = 0.0f;
                    if ((unsigned)m < 48u) {
                        const float* ap = x2base + ca * CSTRIDE + m;
                        a.x = ap[0]; a.y = ap[CSTRIDE];
                    }
                    A[j] = a;
                }
#pragma unroll
                for (int t = 0; t < 3; ++t) {
                    acc[2 * t] = __builtin_amdgcn_wmma_f32_16x16x4_f32(
                        false, A[t], false, Breg[t][i], (short)0, acc[2 * t], false, false);
                    acc[2 * t + 1] = __builtin_amdgcn_wmma_f32_16x16x4_f32(
                        false, A[t + 1], false, Breg[t][i], (short)0, acc[2 * t + 1], false, false);
                }
            }
        }

#pragma unroll
        for (int t = 0; t < 3; ++t) {
#pragma unroll
            for (int r = 0; r < 8; ++r) {
                G[wave][0][r + half * 8][lid] = acc[2 * t][r];
                G[wave][1][r + half * 8][lid] = acc[2 * t + 1][r];
            }
            asm volatile("s_wait_dscnt 0x0" ::: "memory");

            const int obase = (p * 9) * CSTRIDE + hwoff + t * 16 + lid;
#pragma unroll
            for (int dkp = 0; dkp < 5; ++dkp) {
                const int dk = dkp * 2 + half;
                if (dk < 9) {
                    const int idx = lid + dk;
                    float v = (idx < 16) ? G[wave][0][idx][lid]
                                         : G[wave][1][idx - 16][lid];
                    out[obase + dk * CSTRIDE] = v * (1.0f / 32.0f);
                }
            }
            asm volatile("s_wait_dscnt 0x0" ::: "memory");
        }
    }
}

extern "C" void kernel_launch(void* const* d_in, const int* in_sizes, int n_in,
                              void* d_out, int out_size, void* d_ws, size_t ws_size,
                              hipStream_t stream) {
    const float* x1 = (const float*)d_in[0];
    const float* x2 = (const float*)d_in[1];
    float* out = (float*)d_out;

    const size_t need = (size_t)4 * TPLANE * sizeof(_Float16);  // 28.3 MB

    if (ws_size >= need) {
        _Float16* ws = (_Float16*)d_ws;
        hipLaunchKernelGGL(corr_split_kernel, dim3(NPOS / 256, 2), dim3(256), 0,
                           stream, x1, x2, ws);
        hipLaunchKernelGGL(corr3d_wmma_f16_kernel, dim3(48 * 48), dim3(256), 0,
                           stream, ws, out);
    } else {
        hipLaunchKernelGGL(corr3d_wmma_f32_kernel, dim3(48 * 48), dim3(256), 0,
                           stream, x1, x2, out);
    }
}
